// TemporalGCN3_28724741276163
// MI455X (gfx1250) — matrix operand order repaired and checked by reading, not compiled, not run on hardware
//
#include <hip/hip_runtime.h>
#include <hip/hip_bf16.h>

// ---- problem constants (fixed by reference) ----
#define T_STEPS 4
#define NNODES  20000
#define NEDGES  160000
#define INDIM   64
#define DDIM    128
#define GDIM    384      // 3*D
#define NLAYERS 2

typedef __attribute__((ext_vector_type(16))) _Float16 v16h;
typedef __attribute__((ext_vector_type(8)))  float    v8f;

union V16HU { v16h v; uint4 u[2]; };

__device__ __forceinline__ unsigned flip_f32(float f) {
  unsigned u = __float_as_uint(f);
  return (u & 0x80000000u) ? ~u : (u | 0x80000000u);
}
__device__ __forceinline__ float unflip_f32(unsigned u) {
  return __uint_as_float((u & 0x80000000u) ? (u & 0x7fffffffu) : ~u);
}
__device__ __forceinline__ float sigmoidf_(float x) { return 1.0f / (1.0f + __expf(-x)); }

// ------------------------------------------------------------------
// Generic cast f32 -> f16, optionally transposing into Bt[Nc][K]
// tr==0: dst[idx] = src[idx]           (total = K*Nc elements)
// tr==1: dst[o*K+k] = src[k*Nc+o]      (src is K x Nc row-major)
// ------------------------------------------------------------------
__global__ __launch_bounds__(256) void cast_wt(const float* __restrict__ src,
                                               _Float16* __restrict__ dst,
                                               int total, int K, int Nc, int tr) {
  int idx = blockIdx.x * 256 + threadIdx.x;
  if (idx >= total) return;
  float v;
  if (tr) {
    int o = idx / K, k = idx - o * K;
    v = src[(size_t)k * Nc + o];
  } else {
    v = src[idx];
  }
  dst[idx] = (_Float16)v;
}

// ------------------------------------------------------------------
// WMMA GEMM: C[M x Nc] = A[M x K] (f16, row-major) * B + bias
// B supplied as Bt[Nc x K] f16 row-major (i.e. B column-major).
// One wave per 16x16 output tile; K-loop of v_wmma_f32_16x16x32_f16.
// Fragment loads match gfx1250 VGPR layouts:
//   A: lane(0-15)=row M, elems 0-7 <- K[kb..kb+7], 8-15 <- K[kb+16..kb+23],
//      kb = k0 + 8*(lane>>4)
//   B: lane holds column (lane&15), same K order as A (kb = k0 + 8*(lane>>4))
// ------------------------------------------------------------------
__global__ __launch_bounds__(256) void wmma_gemm(const _Float16* __restrict__ A,
                                                 const _Float16* __restrict__ Bt,
                                                 const float* __restrict__ bias,
                                                 float* __restrict__ C,
                                                 int M, int Nc, int K) {
  int wave = blockIdx.x * (256 >> 5) + (threadIdx.x >> 5);
  int tilesN = Nc >> 4;
  int tM = wave / tilesN;
  int tN = wave - tM * tilesN;
  if (tM * 16 >= M) return;                 // uniform per wave -> EXEC stays full
  int lane = threadIdx.x & 31;
  int hsel = lane >> 4;                     // 0 or 1
  int l16  = lane & 15;

  const _Float16* arow = A  + (size_t)(tM * 16 + l16) * K + hsel * 8;
  const _Float16* brow = Bt + (size_t)(tN * 16 + l16) * K + hsel * 8;

  v8f acc = {};
  for (int k0 = 0; k0 < K; k0 += 32) {
    V16HU af, bf;
    af.u[0] = *reinterpret_cast<const uint4*>(arow + k0);
    af.u[1] = *reinterpret_cast<const uint4*>(arow + k0 + 16);
    bf.u[0] = *reinterpret_cast<const uint4*>(brow + k0);
    bf.u[1] = *reinterpret_cast<const uint4*>(brow + k0 + 16);
    acc = __builtin_amdgcn_wmma_f32_16x16x32_f16(false, af.v, false, bf.v,
                                                 (short)0, acc, false, false);
  }
  int col = tN * 16 + l16;
  float b = bias ? bias[col] : 0.0f;
  float* crow = C + (size_t)(tM * 16 + hsel * 8) * Nc + col;
#pragma unroll
  for (int r = 0; r < 8; ++r) crow[(size_t)r * Nc] = acc[r] + b;
}

// ------------------------------------------------------------------
// GRU gate math (torch order r,z,n); biases already folded into gi/gh.
// ------------------------------------------------------------------
__global__ __launch_bounds__(256) void gru_gates(const float* __restrict__ gi,
                                                 const float* __restrict__ gh,
                                                 float* __restrict__ h32,
                                                 _Float16* __restrict__ h16) {
  int idx = blockIdx.x * 256 + threadIdx.x;           // n*128 + d, exact grid
  int n = idx >> 7, d = idx & 127;
  const float* gin = gi + (size_t)n * GDIM;
  const float* ghn = gh + (size_t)n * GDIM;
  float r  = sigmoidf_(gin[d] + ghn[d]);
  float z  = sigmoidf_(gin[DDIM + d] + ghn[DDIM + d]);
  float nn = tanhf(gin[2 * DDIM + d] + r * ghn[2 * DDIM + d]);
  float h  = h32[idx];
  float hn = (1.0f - z) * nn + z * h;
  h32[idx] = hn;
  h16[idx] = (_Float16)hn;
}

// ------------------------------------------------------------------
// Edge kernels for D=128 attention: one wave32 per edge, 4 dims/lane.
// ------------------------------------------------------------------
__global__ __launch_bounds__(256) void edge_alpha(const int* __restrict__ src,
                                                  const int* __restrict__ dst,
                                                  const float* __restrict__ attr,
                                                  const float* __restrict__ q,
                                                  const float* __restrict__ kk,
                                                  const float* __restrict__ We,
                                                  float* __restrict__ alpha,
                                                  unsigned* __restrict__ amax,
                                                  int E, float scale) {
  int e = blockIdx.x * 8 + (threadIdx.x >> 5);
  if (e >= E) return;
  int lane = threadIdx.x & 31;
  int s = src[e], d = dst[e];
  float a = attr[e];
  float4 qq = reinterpret_cast<const float4*>(q  + (size_t)d * DDIM)[lane];
  float4 kv = reinterpret_cast<const float4*>(kk + (size_t)s * DDIM)[lane];
  float4 wv = reinterpret_cast<const float4*>(We)[lane];
  float p = qq.x * (kv.x + a * wv.x) + qq.y * (kv.y + a * wv.y) +
            qq.z * (kv.z + a * wv.z) + qq.w * (kv.w + a * wv.w);
#pragma unroll
  for (int o = 16; o; o >>= 1) p += __shfl_xor(p, o, 32);
  if (lane == 0) {
    p *= scale;
    alpha[e] = p;
    atomicMax(amax + d, flip_f32(p));
  }
}

__global__ __launch_bounds__(256) void edge_expsum(const int* __restrict__ dst,
                                                   const float* __restrict__ alpha,
                                                   const unsigned* __restrict__ amax,
                                                   float* __restrict__ exb,
                                                   float* __restrict__ denom, int E) {
  int e = blockIdx.x * 256 + threadIdx.x;
  if (e >= E) return;
  int d = dst[e];
  float ex = __expf(alpha[e] - unflip_f32(amax[d]));
  exb[e] = ex;
  atomicAdd(denom + d, ex);
}

__global__ __launch_bounds__(256) void edge_scatter(const int* __restrict__ src,
                                                    const int* __restrict__ dst,
                                                    const float* __restrict__ attr,
                                                    const float* __restrict__ v,
                                                    const float* __restrict__ We,
                                                    const float* __restrict__ exb,
                                                    const float* __restrict__ denom,
                                                    float* __restrict__ agg, int E) {
  int e = blockIdx.x * 8 + (threadIdx.x >> 5);
  if (e >= E) return;
  int lane = threadIdx.x & 31;
  int s = src[e], d = dst[e];
  float a = attr[e];
  float attn = exb[e] / (denom[d] + 1e-16f);
  float4 vv = reinterpret_cast<const float4*>(v + (size_t)s * DDIM)[lane];
  float4 wv = reinterpret_cast<const float4*>(We)[lane];
  float* ag = agg + (size_t)d * DDIM + lane * 4;
  atomicAdd(ag + 0, (vv.x + a * wv.x) * attn);
  atomicAdd(ag + 1, (vv.y + a * wv.y) * attn);
  atomicAdd(ag + 2, (vv.z + a * wv.z) * attn);
  atomicAdd(ag + 3, (vv.w + a * wv.w) * attn);
}

__global__ __launch_bounds__(256) void conv_combine(const float* __restrict__ agg,
                                                    const float* __restrict__ skip,
                                                    _Float16* __restrict__ h16out,
                                                    float* __restrict__ hmean,
                                                    int addMean) {
  int idx = blockIdx.x * 256 + threadIdx.x;           // exact grid N*128
  float o = agg[idx] + skip[idx];
  o = (o > 0.0f) ? o : 0.01f * o;                     // leaky_relu
  h16out[idx] = (_Float16)o;
  if (addMean) hmean[idx] += 0.25f * o;
}

// ------------------------------------------------------------------
// Final D->1 conv: per-node projections then scalar edge softmax.
// ------------------------------------------------------------------
__global__ __launch_bounds__(256) void fin_proj(const float* __restrict__ hmean,
                                                const float* __restrict__ Wq, const float* __restrict__ bq,
                                                const float* __restrict__ Wk, const float* __restrict__ bk,
                                                const float* __restrict__ Wv, const float* __restrict__ bv,
                                                const float* __restrict__ Ws, const float* __restrict__ bs,
                                                float* __restrict__ q1, float* __restrict__ k1,
                                                float* __restrict__ v1, float* __restrict__ s1, int N) {
  int n = blockIdx.x * 8 + (threadIdx.x >> 5);
  if (n >= N) return;
  int lane = threadIdx.x & 31;
  float4 h = reinterpret_cast<const float4*>(hmean + (size_t)n * DDIM)[lane];
  float4 w;
  w = reinterpret_cast<const float4*>(Wq)[lane];
  float pq = h.x * w.x + h.y * w.y + h.z * w.z + h.w * w.w;
  w = reinterpret_cast<const float4*>(Wk)[lane];
  float pk = h.x * w.x + h.y * w.y + h.z * w.z + h.w * w.w;
  w = reinterpret_cast<const float4*>(Wv)[lane];
  float pv = h.x * w.x + h.y * w.y + h.z * w.z + h.w * w.w;
  w = reinterpret_cast<const float4*>(Ws)[lane];
  float ps = h.x * w.x + h.y * w.y + h.z * w.z + h.w * w.w;
#pragma unroll
  for (int o = 16; o; o >>= 1) {
    pq += __shfl_xor(pq, o, 32);
    pk += __shfl_xor(pk, o, 32);
    pv += __shfl_xor(pv, o, 32);
    ps += __shfl_xor(ps, o, 32);
  }
  if (lane == 0) {
    q1[n] = pq + bq[0];
    k1[n] = pk + bk[0];
    v1[n] = pv + bv[0];
    s1[n] = ps + bs[0];
  }
}

__global__ __launch_bounds__(256) void fin_alpha(const int* __restrict__ src,
                                                 const int* __restrict__ dst,
                                                 const float* __restrict__ attr,
                                                 const float* __restrict__ q1,
                                                 const float* __restrict__ k1,
                                                 const float* __restrict__ We,
                                                 float* __restrict__ alpha,
                                                 unsigned* __restrict__ amax, int E) {
  int e = blockIdx.x * 256 + threadIdx.x;
  if (e >= E) return;
  int s = src[e], d = dst[e];
  float al = q1[d] * (k1[s] + attr[e] * We[0]);      // d_out=1 -> scale 1.0
  alpha[e] = al;
  atomicMax(amax + d, flip_f32(al));
}

__global__ __launch_bounds__(256) void fin_scatter(const int* __restrict__ src,
                                                   const int* __restrict__ dst,
                                                   const float* __restrict__ attr,
                                                   const float* __restrict__ v1,
                                                   const float* __restrict__ We,
                                                   const float* __restrict__ exb,
                                                   const float* __restrict__ denom,
                                                   float* __restrict__ agg1, int E) {
  int e = blockIdx.x * 256 + threadIdx.x;
  if (e >= E) return;
  int s = src[e], d = dst[e];
  float attn = exb[e] / (denom[d] + 1e-16f);
  atomicAdd(agg1 + d, (v1[s] + attr[e] * We[0]) * attn);
}

__global__ __launch_bounds__(256) void fin_out(const float* __restrict__ agg1,
                                               const float* __restrict__ s1,
                                               float* __restrict__ out, int N) {
  int n = blockIdx.x * 256 + threadIdx.x;
  if (n >= N) return;
  out[n] = agg1[n] + s1[n];
}

// ------------------------------------------------------------------
extern "C" void kernel_launch(void* const* d_in, const int* in_sizes, int n_in,
                              void* d_out, int out_size, void* d_ws, size_t ws_size,
                              hipStream_t stream) {
  const float* x_seq  = (const float*)d_in[0];
  const int*   ei_seq = (const int*)d_in[1];
  const float* ea_seq = (const float*)d_in[2];
  const float* W_ih   = (const float*)d_in[3];
  const float* W_hh   = (const float*)d_in[4];
  const float* b_ih   = (const float*)d_in[5];
  const float* b_hh   = (const float*)d_in[6];
  const float* cWq    = (const float*)d_in[7];
  const float* cbq    = (const float*)d_in[8];
  const float* cWk    = (const float*)d_in[9];
  const float* cbk    = (const float*)d_in[10];
  const float* cWv    = (const float*)d_in[11];
  const float* cbv    = (const float*)d_in[12];
  const float* cWe    = (const float*)d_in[13];
  const float* cWs    = (const float*)d_in[14];
  const float* cbs    = (const float*)d_in[15];
  const float* oWq    = (const float*)d_in[16];
  const float* obq    = (const float*)d_in[17];
  const float* oWk    = (const float*)d_in[18];
  const float* obk    = (const float*)d_in[19];
  const float* oWv    = (const float*)d_in[20];
  const float* obv    = (const float*)d_in[21];
  const float* oWe    = (const float*)d_in[22];
  const float* oWs    = (const float*)d_in[23];
  const float* obs    = (const float*)d_in[24];

  // ---- workspace bump allocator (256B aligned) ----
  char* base = (char*)d_ws;
  size_t off = 0;
  auto alloc = [&](size_t bytes) -> char* {
    off = (off + 255) & ~(size_t)255;
    char* p = base + off;
    off += bytes;
    return p;
  };
  const size_t ND4 = (size_t)NNODES * DDIM * 4;       // 10.24 MB
  const size_t ND2 = (size_t)NNODES * DDIM * 2;

  _Float16* wih16 = (_Float16*)alloc((size_t)GDIM * INDIM * 2);
  _Float16* whh16 = (_Float16*)alloc((size_t)GDIM * DDIM * 2);
  _Float16* wq16  = (_Float16*)alloc((size_t)NLAYERS * DDIM * DDIM * 2);
  _Float16* wk16  = (_Float16*)alloc((size_t)NLAYERS * DDIM * DDIM * 2);
  _Float16* wv16  = (_Float16*)alloc((size_t)NLAYERS * DDIM * DDIM * 2);
  _Float16* ws16  = (_Float16*)alloc((size_t)NLAYERS * DDIM * DDIM * 2);
  _Float16* x16   = (_Float16*)alloc((size_t)T_STEPS * NNODES * INDIM * 2);
  float*    h32   = (float*)alloc(ND4);
  _Float16* gh16  = (_Float16*)alloc(ND2);            // GRU hidden as f16
  _Float16* ch16  = (_Float16*)alloc(ND2);            // conv layer activation f16
  char*     bufA  = alloc((size_t)NNODES * GDIM * 4); // gi  | q,k,v
  char*     bufB  = alloc((size_t)NNODES * GDIM * 4); // gh  | skip,agg,alpha,ex,amax,denom
  float*    hmean = (float*)alloc(ND4);
  float*    q1    = (float*)alloc((size_t)NNODES * 4);
  float*    k1    = (float*)alloc((size_t)NNODES * 4);
  float*    v1    = (float*)alloc((size_t)NNODES * 4);
  float*    s1    = (float*)alloc((size_t)NNODES * 4);
  float*    agg1  = (float*)alloc((size_t)NNODES * 4);
  float*    den1  = (float*)alloc((size_t)NNODES * 4);
  unsigned* amax1 = (unsigned*)alloc((size_t)NNODES * 4);

  float* gi    = (float*)bufA;
  float* qf    = (float*)bufA;
  float* kf    = (float*)(bufA + ND4);
  float* vf    = (float*)(bufA + 2 * ND4);
  float* gh    = (float*)bufB;
  float* skipf = (float*)bufB;
  float* aggf  = (float*)(bufB + ND4);
  float* alpha = (float*)(bufB + 2 * ND4);
  float* exb   = (float*)(bufB + 2 * ND4 + (size_t)NEDGES * 4);
  unsigned* amax = (unsigned*)(bufB + 2 * ND4 + (size_t)NEDGES * 8);
  float* denom = (float*)(bufB + 2 * ND4 + (size_t)NEDGES * 8 + (size_t)NNODES * 4);

  const float SCALE_D = 0.08838834764831844f;         // 1/sqrt(128)

  // ---- weight / input casts (every call; deterministic) ----
  cast_wt<<<(GDIM * INDIM + 255) / 256, 256, 0, stream>>>(W_ih, wih16, GDIM * INDIM, INDIM, GDIM, 0);
  cast_wt<<<(GDIM * DDIM + 255) / 256, 256, 0, stream>>>(W_hh, whh16, GDIM * DDIM, DDIM, GDIM, 0);
  for (int l = 0; l < NLAYERS; ++l) {
    int n = DDIM * DDIM, g = (n + 255) / 256;
    size_t o = (size_t)l * DDIM * DDIM;
    cast_wt<<<g, 256, 0, stream>>>(cWq + o, wq16 + o, n, DDIM, DDIM, 1);
    cast_wt<<<g, 256, 0, stream>>>(cWk + o, wk16 + o, n, DDIM, DDIM, 1);
    cast_wt<<<g, 256, 0, stream>>>(cWv + o, wv16 + o, n, DDIM, DDIM, 1);
    cast_wt<<<g, 256, 0, stream>>>(cWs + o, ws16 + o, n, DDIM, DDIM, 1);
  }
  {
    int n = T_STEPS * NNODES * INDIM;
    cast_wt<<<(n + 255) / 256, 256, 0, stream>>>(x_seq, x16, n, 1, n, 0);
  }
  hipMemsetAsync(h32, 0, ND4, stream);
  hipMemsetAsync(gh16, 0, ND2, stream);
  hipMemsetAsync(hmean, 0, ND4, stream);

  const int tilesM = NNODES / 16;                     // 1250
  const int gemmG_gru  = (tilesM * (GDIM / 16) + 7) / 8;
  const int gemmG_conv = (tilesM * (DDIM / 16) + 7) / 8;
  const int gridNd     = NNODES * DDIM / 256;         // 10000
  const int gridEw     = NEDGES / 8;                  // wave-per-edge
  const int gridEt     = NEDGES / 256;                // thread-per-edge

  for (int t = 0; t < T_STEPS; ++t) {
    const int*   ei = ei_seq + (size_t)t * 2 * NEDGES;
    const int*   esrc = ei;
    const int*   edst = ei + NEDGES;
    const float* ea = ea_seq + (size_t)t * NEDGES;

    // GRU step: gi = x_t @ W_ih^T + b_ih ; gh = h @ W_hh^T + b_hh
    wmma_gemm<<<gemmG_gru, 256, 0, stream>>>(x16 + (size_t)t * NNODES * INDIM,
                                             wih16, b_ih, gi, NNODES, GDIM, INDIM);
    wmma_gemm<<<gemmG_gru, 256, 0, stream>>>(gh16, whh16, b_hh, gh, NNODES, GDIM, DDIM);
    gru_gates<<<gridNd, 256, 0, stream>>>(gi, gh, h32, gh16);

    _Float16* cur = gh16;
    for (int l = 0; l < NLAYERS; ++l) {
      size_t wo = (size_t)l * DDIM * DDIM;
      const float* We = cWe + (size_t)l * DDIM;
      wmma_gemm<<<gemmG_conv, 256, 0, stream>>>(cur, wq16 + wo, cbq + l * DDIM, qf, NNODES, DDIM, DDIM);
      wmma_gemm<<<gemmG_conv, 256, 0, stream>>>(cur, wk16 + wo, cbk + l * DDIM, kf, NNODES, DDIM, DDIM);
      wmma_gemm<<<gemmG_conv, 256, 0, stream>>>(cur, wv16 + wo, cbv + l * DDIM, vf, NNODES, DDIM, DDIM);
      wmma_gemm<<<gemmG_conv, 256, 0, stream>>>(cur, ws16 + wo, cbs + l * DDIM, skipf, NNODES, DDIM, DDIM);

      hipMemsetAsync(aggf, 0, ND4, stream);
      hipMemsetAsync(amax, 0, (size_t)NNODES * 8, stream);   // amax + denom contiguous

      edge_alpha<<<gridEw, 256, 0, stream>>>(esrc, edst, ea, qf, kf, We, alpha, amax, NEDGES, SCALE_D);
      edge_expsum<<<gridEt, 256, 0, stream>>>(edst, alpha, amax, exb, denom, NEDGES);
      edge_scatter<<<gridEw, 256, 0, stream>>>(esrc, edst, ea, vf, We, exb, denom, aggf, NEDGES);
      conv_combine<<<gridNd, 256, 0, stream>>>(aggf, skipf, ch16, hmean, (l == NLAYERS - 1) ? 1 : 0);
      cur = ch16;
    }
  }

  // ---- final TransformerConv (D -> 1) on mean, last timestep's graph ----
  {
    const int*   ei = ei_seq + (size_t)(T_STEPS - 1) * 2 * NEDGES;
    const int*   esrc = ei;
    const int*   edst = ei + NEDGES;
    const float* ea = ea_seq + (size_t)(T_STEPS - 1) * NEDGES;

    fin_proj<<<(NNODES + 7) / 8, 256, 0, stream>>>(hmean, oWq, obq, oWk, obk, oWv, obv, oWs, obs,
                                                   q1, k1, v1, s1, NNODES);
    hipMemsetAsync(agg1, 0, (size_t)NNODES * 4, stream);
    hipMemsetAsync(den1, 0, (size_t)NNODES * 4, stream);
    hipMemsetAsync(amax1, 0, (size_t)NNODES * 4, stream);
    fin_alpha<<<gridEt, 256, 0, stream>>>(esrc, edst, ea, q1, k1, oWe, alpha, amax1, NEDGES);
    edge_expsum<<<gridEt, 256, 0, stream>>>(edst, alpha, amax1, exb, den1, NEDGES);
    fin_scatter<<<gridEt, 256, 0, stream>>>(esrc, edst, ea, v1, oWe, exb, den1, agg1, NEDGES);
    fin_out<<<(NNODES + 255) / 256, 256, 0, stream>>>(agg1, s1, (float*)d_out, NNODES);
  }
}
